// TwoStageAtlasNetGenerator_52527450030145
// MI455X (gfx1250) — matrix operand order repaired and checked by reading, not compiled, not run on hardware
//
#include <hip/hip_runtime.h>
#include <hip/hip_bf16.h>
#include <math.h>

// ---------- types ----------
typedef __attribute__((ext_vector_type(16))) _Float16 v16h;
typedef __attribute__((ext_vector_type(8)))  _Float16 v8h;
typedef __attribute__((ext_vector_type(8)))  float    v8f;

#define NEG_SLOPE 0.2f
#define RADIUS2   0.01f
#define NS        64
#define NB        2
#define N1V       1024
#define N2V       2048

// ---------- workspace layout (bytes) ----------
#define OFF_W1B   (size_t)0          // 512x512 f16 frags: 512KB
#define OFF_W1C   (size_t)524288
#define OFF_W2A   (size_t)1048576    // 512x1536: 1.5MB
#define OFF_W2B   (size_t)2621440
#define OFF_W2C   (size_t)3145728
#define OFF_WX1   (size_t)3670016    // 16x512 (rows>=3 zero): 16KB
#define OFF_WX2   (size_t)3686400
#define OFF_BASE1 (size_t)3702784    // 2x512 f32
#define OFF_H2    (size_t)3706880    // [b][n][c] f16: 4MB
#define OFF_LOCAL (size_t)7901184    // [b][n][c] f16: 4MB
#define OFF_GLOB  (size_t)12095488   // [b][c] f16
#define OFF_NBR   (size_t)12097536   // [b][n][64] u16: 512KB

__device__ __forceinline__ float leakyf(float x){ return x > 0.f ? x : NEG_SLOPE * x; }
__device__ __forceinline__ float sigmf(float x){ return 1.0f / (1.0f + __expf(-x)); }

__device__ __forceinline__ v8f wmma_f16(v16h a, v16h b, v8f c) {
  return __builtin_amdgcn_wmma_f32_16x16x32_f16(false, a, false, b, (short)0, c, false, false);
}

// =====================================================================
// Weight prep: pack f32 O x C (row-major) into WMMA A-fragment tiles.
// Tile (ot,kt) = 16 rows x 32 K. Contiguous-K A layout used by this file:
//   lane L (m=L&15, g=L>>4): halves 0..15 -> K = g*16 + h
// Tile storage: 32 lanes x 16 halves contiguous (1KB / tile),
// tile index = ot*numKT + kt.  Rows >= realO are zero (Wx pad 3->16).
// =====================================================================
__global__ void prep_weight_kernel(const float* __restrict__ src,
                                   _Float16* __restrict__ dst,
                                   int numOT, int realO, int C) {
  int numKT = C >> 5;
  int total = numOT * numKT * 32;
  int id = blockIdx.x * 256 + threadIdx.x;
  if (id >= total) return;
  int tile = id >> 5, L = id & 31;
  int ot = tile / numKT, kt = tile - ot * numKT;
  int m = L & 15, g = L >> 4;
  int row = ot * 16 + m;
  v16h out;
#pragma unroll
  for (int h = 0; h < 16; ++h) {
    int K = g * 16 + h;
    float v = (row < realO) ? src[(size_t)row * C + kt * 32 + K] : 0.f;
    out[h] = (_Float16)v;
  }
  *(v16h*)(dst + (size_t)tile * 512 + L * 16) = out;
}

// base1[b][o] = sum_c W1a[o, c] * z[b, c] + b1a[o]   (c in 0..511)
__global__ void base1_kernel(const float* __restrict__ W1a,
                             const float* __restrict__ z,
                             const float* __restrict__ b1a,
                             float* __restrict__ base1) {
  int o = blockIdx.x * 256 + threadIdx.x;
  int b = blockIdx.y;
  const float* wr = W1a + (size_t)o * 514;
  const float* zr = z + b * 512;
  float acc = b1a[o];
  for (int c = 0; c < 512; ++c) acc += wr[c] * zr[c];
  base1[b * 512 + o] = acc;
}

// =====================================================================
// GEMM layer: out(512 x 16*NUMPT) = leaky(W(512 x 32*numKT) @ in + bias)
// A frags from global (prepacked), B frags from LDS (fragment layout:
// lane l holds column n=l%16, K=(l/16)*16 + half -> 32B/lane/tile).
// D epilogue: lane's 8 rows are 8 contiguous halves of one B-frag
// lane-row of the next layer -> single 16B LDS store per lane/tile.
// =====================================================================
template<int NUMPT>
__device__ __forceinline__ void gemm_layer(const _Float16* __restrict__ wfrag,
                                           const float* __restrict__ bias,
                                           const _Float16* __restrict__ inBuf,
                                           _Float16* __restrict__ outBuf,
                                           int numKT, int w, int lane,
                                           _Float16* __restrict__ plainOut,
                                           int writePlain, size_t plainRowStart) {
  const int nn = lane & 15;
  const int hi = (lane < 16) ? 0 : 8;
#pragma unroll
  for (int oi = 0; oi < 4; ++oi) {
    int ot = w * 4 + oi;
    v8f acc[NUMPT] = {};
    for (int kt = 0; kt < numKT; ++kt) {
      v16h a = *(const v16h*)(wfrag + ((size_t)(ot * numKT + kt)) * 512 + lane * 16);
#pragma unroll
      for (int pt = 0; pt < NUMPT; ++pt) {
        v16h bb = *(const v16h*)(inBuf + (kt * NUMPT + pt) * 512 + lane * 16);
        acc[pt] = wmma_f16(a, bb, acc[pt]);
      }
    }
    union { float4 v[2]; float f[8]; } bu;
    const float4* bp = (const float4*)(bias + ot * 16 + hi);
    bu.v[0] = bp[0]; bu.v[1] = bp[1];
#pragma unroll
    for (int pt = 0; pt < NUMPT; ++pt) {
      v8h hx;
#pragma unroll
      for (int r = 0; r < 8; ++r) hx[r] = (_Float16)leakyf(acc[pt][r] + bu.f[r]);
      // next-layer B-frag: kt' = ot>>1, K base = (ot&1)*16
      *(v8h*)(outBuf + ((ot >> 1) * NUMPT + pt) * 512 + ((ot & 1) * 16 + nn) * 16 + hi) = hx;
      if (writePlain)
        *(v8h*)(plainOut + plainRowStart + (size_t)(pt * 16 + nn) * 512 + ot * 16 + hi) = hx;
    }
  }
}

// 16-row head GEMM (rows >= 3 zero-padded in the frag). One wave per pt.
template<int NUMPT>
__device__ __forceinline__ v8f gemm_head(const _Float16* __restrict__ wxF,
                                         const _Float16* __restrict__ inBuf,
                                         int numKT, int pt, int lane) {
  v8f acc = {};
  for (int kt = 0; kt < numKT; ++kt) {
    v16h a  = *(const v16h*)(wxF + (size_t)kt * 512 + lane * 16);
    v16h bb = *(const v16h*)(inBuf + (kt * NUMPT + pt) * 512 + lane * 16);
    acc = wmma_f16(a, bb, acc);
  }
  return acc;
}

// =====================================================================
// Stage-1: 64 points / workgroup (8 waves).  LDS: bufA(64KB) + bufB(64KB)
// =====================================================================
__global__ void stage1_kernel(const float* __restrict__ base1,
                              const float* __restrict__ W1a,
                              const _Float16* __restrict__ w1bF, const float* __restrict__ b1b,
                              const _Float16* __restrict__ w1cF, const float* __restrict__ b1c,
                              const _Float16* __restrict__ wx1F, const float* __restrict__ bx1,
                              float* __restrict__ pts, _Float16* __restrict__ hOut,
                              int n, int side, int writeH) {
  extern __shared__ char smem[];
  _Float16* bufA = (_Float16*)smem;            // 512 x 64 (16 kt x 4 pt frags)
  _Float16* bufB = (_Float16*)(smem + 65536);
  const int tid = threadIdx.x;
  const int w = tid >> 5, lane = tid & 31;
  const int b = blockIdx.y;
  const int p0 = blockIdx.x * 64;
  const float inv = 1.0f / (float)(side - 1);

  // layer 0: a0 = leaky(base + W1a[:,512]*lx + W1a[:,513]*ly), packed as B frags
  for (int id = tid; id < 2048; id += 256) {
    int kt = id >> 7, rem = id & 127;
    int pt = rem >> 5, L = rem & 31;
    int g = L >> 4, nn = L & 15;
    int p = p0 + pt * 16 + nn;
    float lx = (float)(p / side) * inv;
    float ly = (float)(p % side) * inv;
    int c0 = kt * 32 + g * 16;
    v16h out;
#pragma unroll
    for (int h = 0; h < 16; ++h) {
      int c = c0 + h;
      float v = base1[b * 512 + c]
              + W1a[(size_t)c * 514 + 512] * lx
              + W1a[(size_t)c * 514 + 513] * ly;
      out[h] = (_Float16)leakyf(v);
    }
    *(v16h*)(bufA + (kt * 4 + pt) * 512 + L * 16) = out;
  }
  __syncthreads();

  gemm_layer<4>(w1bF, b1b, bufA, bufB, 16, w, lane, (_Float16*)0, 0, 0);
  __syncthreads();
  gemm_layer<4>(w1cF, b1c, bufB, bufA, 16, w, lane, hOut, writeH,
                ((size_t)b * n + p0) * 512);
  __syncthreads();

  if (w < 4) {
    v8f d = gemm_head<4>(wx1F, bufA, 16, w, lane);
    if (lane < 16) {
      int p = p0 + w * 16 + lane;
#pragma unroll
      for (int r = 0; r < 3; ++r)
        pts[((size_t)b * n + p) * 3 + r] = sigmf(d[r] + bx1[r]) - 0.5f;
    }
  }
}

// glob[b][c] = max_n h2[b][n][c]
__global__ void glob_kernel(const _Float16* __restrict__ h2, _Float16* __restrict__ glob) {
  int c = blockIdx.x * 256 + threadIdx.x;
  int b = blockIdx.y;
  float m = -3.0e38f;
  for (int nn = 0; nn < N2V; ++nn)
    m = fmaxf(m, (float)h2[((size_t)b * N2V + nn) * 512 + c]);
  glob[b * 512 + c] = (_Float16)m;
}

// First NS ascending j with |pi-pj|^2 < R2 (self always qualifies); pad with first.
__global__ void nbr_kernel(const float* __restrict__ fine, unsigned short* __restrict__ nbr) {
  __shared__ float sx[256], sy[256], sz[256];
  int i = blockIdx.x * 256 + threadIdx.x;
  int b = blockIdx.y;
  const float* pi = fine + ((size_t)b * N2V + i) * 3;
  float px = pi[0], py = pi[1], pz = pi[2];
  unsigned short* row = nbr + ((size_t)b * N2V + i) * NS;
  int cnt = 0; unsigned short first = 0;
  for (int j0 = 0; j0 < N2V; j0 += 256) {
    __syncthreads();
    {
      const float* pj = fine + ((size_t)b * N2V + j0 + threadIdx.x) * 3;
      sx[threadIdx.x] = pj[0]; sy[threadIdx.x] = pj[1]; sz[threadIdx.x] = pj[2];
    }
    __syncthreads();
    for (int jj = 0; jj < 256; ++jj) {
      float dx = px - sx[jj], dy = py - sy[jj], dz = pz - sz[jj];
      float d2 = dx * dx + dy * dy + dz * dz;
      if (d2 < RADIUS2 && cnt < NS) {
        unsigned short j = (unsigned short)(j0 + jj);
        if (cnt == 0) first = j;
        row[cnt] = j;
        cnt++;
      }
    }
  }
  for (int s = cnt; s < NS; ++s) row[s] = first;
}

// local[b][i][c] = max over 64 neighbors of h2[b][j][c]
__global__ void local_kernel(const _Float16* __restrict__ h2,
                             const unsigned short* __restrict__ nbr,
                             _Float16* __restrict__ localf) {
  int i = blockIdx.x, b = blockIdx.y;
  const unsigned short* row = nbr + ((size_t)b * N2V + i) * NS;
  int c = threadIdx.x;
  float m0 = -3.0e38f, m1 = -3.0e38f;
  for (int s = 0; s < NS; ++s) {
    int j = row[s];
    const _Float16* hr = h2 + ((size_t)b * N2V + j) * 512;
    m0 = fmaxf(m0, (float)hr[c]);
    m1 = fmaxf(m1, (float)hr[c + 256]);
  }
  _Float16* lr = localf + ((size_t)b * N2V + i) * 512;
  lr[c] = (_Float16)m0;
  lr[c + 256] = (_Float16)m1;
}

// =====================================================================
// Stage-2: 32 points / WG. LDS: bufIn 96KB (1536x32) + 2 x 32KB ping-pong.
// =====================================================================
__global__ void stage2_kernel(const _Float16* __restrict__ h2,
                              const _Float16* __restrict__ localf,
                              const _Float16* __restrict__ glob,
                              const _Float16* __restrict__ w2aF, const float* __restrict__ b2a,
                              const _Float16* __restrict__ w2bF, const float* __restrict__ b2b,
                              const _Float16* __restrict__ w2cF, const float* __restrict__ b2c,
                              const _Float16* __restrict__ wx2F, const float* __restrict__ bx2,
                              const float* __restrict__ fine, float* __restrict__ out2) {
  extern __shared__ char smem[];
  _Float16* bufIn = (_Float16*)smem;                    // 48 kt x 2 pt frags
  _Float16* bufB  = (_Float16*)(smem + 98304);          // 512 x 32
  _Float16* bufC  = (_Float16*)(smem + 98304 + 32768);
  const int tid = threadIdx.x;
  const int w = tid >> 5, lane = tid & 31;
  const int b = blockIdx.y;
  const int p0 = blockIdx.x * 32;

  // layer 0: pack feat = [h2 | local | glob] into B frags (16 contiguous halves per store)
  for (int id = tid; id < 3072; id += 256) {
    int tile = id >> 5, L = id & 31;
    int kt = tile >> 1, pt = tile & 1;
    int g = L >> 4, nn = L & 15;
    int p = p0 + pt * 16 + nn;
    int c0 = kt * 32 + g * 16;
    const _Float16* src;
    if (c0 < 512)       src = h2     + ((size_t)b * N2V + p) * 512 + c0;
    else if (c0 < 1024) src = localf + ((size_t)b * N2V + p) * 512 + (c0 - 512);
    else                src = glob   + b * 512 + (c0 - 1024);
    *(v16h*)(bufIn + tile * 512 + L * 16) = *(const v16h*)src;
  }
  __syncthreads();

  gemm_layer<2>(w2aF, b2a, bufIn, bufB, 48, w, lane, (_Float16*)0, 0, 0);
  __syncthreads();
  gemm_layer<2>(w2bF, b2b, bufB, bufC, 16, w, lane, (_Float16*)0, 0, 0);
  __syncthreads();
  gemm_layer<2>(w2cF, b2c, bufC, bufB, 16, w, lane, (_Float16*)0, 0, 0);
  __syncthreads();

  if (w < 2) {
    v8f d = gemm_head<2>(wx2F, bufB, 16, w, lane);
    if (lane < 16) {
      int p = p0 + w * 16 + lane;
#pragma unroll
      for (int r = 0; r < 3; ++r) {
        float delta = 0.1f * (sigmf(d[r] + bx2[r]) - 0.5f);
        out2[((size_t)b * N2V + p) * 3 + r] = fine[((size_t)b * N2V + p) * 3 + r] + delta;
      }
    }
  }
}

// =====================================================================
extern "C" void kernel_launch(void* const* d_in, const int* in_sizes, int n_in,
                              void* d_out, int out_size, void* d_ws, size_t ws_size,
                              hipStream_t stream) {
  (void)in_sizes; (void)n_in; (void)out_size; (void)ws_size;
  const float* z   = (const float*)d_in[0];
  const float* W1a = (const float*)d_in[1];
  const float* b1a = (const float*)d_in[2];
  const float* W1b = (const float*)d_in[3];
  const float* b1b = (const float*)d_in[4];
  const float* W1c = (const float*)d_in[5];
  const float* b1c = (const float*)d_in[6];
  const float* Wx1 = (const float*)d_in[7];
  const float* bx1 = (const float*)d_in[8];
  const float* W2a = (const float*)d_in[9];
  const float* b2a = (const float*)d_in[10];
  const float* W2b = (const float*)d_in[11];
  const float* b2b = (const float*)d_in[12];
  const float* W2c = (const float*)d_in[13];
  const float* b2c = (const float*)d_in[14];
  const float* Wx2 = (const float*)d_in[15];
  const float* bx2 = (const float*)d_in[16];

  char* ws = (char*)d_ws;
  _Float16* w1bF  = (_Float16*)(ws + OFF_W1B);
  _Float16* w1cF  = (_Float16*)(ws + OFF_W1C);
  _Float16* w2aF  = (_Float16*)(ws + OFF_W2A);
  _Float16* w2bF  = (_Float16*)(ws + OFF_W2B);
  _Float16* w2cF  = (_Float16*)(ws + OFF_W2C);
  _Float16* wx1F  = (_Float16*)(ws + OFF_WX1);
  _Float16* wx2F  = (_Float16*)(ws + OFF_WX2);
  float*    base1 = (float*)   (ws + OFF_BASE1);
  _Float16* h2    = (_Float16*)(ws + OFF_H2);
  _Float16* locf  = (_Float16*)(ws + OFF_LOCAL);
  _Float16* glob  = (_Float16*)(ws + OFF_GLOB);
  unsigned short* nbr = (unsigned short*)(ws + OFF_NBR);

  float* out       = (float*)d_out;
  float* firstPts  = out;           // (2,1024,3)
  float* secondPts = out + 6144;    // (2,2048,3)
  float* finePts   = out + 18432;   // (2,2048,3)

  // weight packing (A-fragment layout)
  prep_weight_kernel<<<64,  256, 0, stream>>>(W1b, w1bF, 32, 512, 512);
  prep_weight_kernel<<<64,  256, 0, stream>>>(W1c, w1cF, 32, 512, 512);
  prep_weight_kernel<<<192, 256, 0, stream>>>(W2a, w2aF, 32, 512, 1536);
  prep_weight_kernel<<<64,  256, 0, stream>>>(W2b, w2bF, 32, 512, 512);
  prep_weight_kernel<<<64,  256, 0, stream>>>(W2c, w2cF, 32, 512, 512);
  prep_weight_kernel<<<2,   256, 0, stream>>>(Wx1, wx1F, 1, 3, 512);
  prep_weight_kernel<<<2,   256, 0, stream>>>(Wx2, wx2F, 1, 3, 512);

  base1_kernel<<<dim3(2, 2), 256, 0, stream>>>(W1a, z, b1a, base1);

  // stage 1, n1=1024 (side=32): first_stage_points
  stage1_kernel<<<dim3(16, 2), 256, 131072, stream>>>(
      base1, W1a, w1bF, b1b, w1cF, b1c, wx1F, bx1,
      firstPts, h2, N1V, 32, 0);

  // stage 1, n2=2048 (side=46): fine_points + h2
  stage1_kernel<<<dim3(32, 2), 256, 131072, stream>>>(
      base1, W1a, w1bF, b1b, w1cF, b1c, wx1F, bx1,
      finePts, h2, N2V, 46, 1);

  glob_kernel <<<dim3(2, 2),    256, 0, stream>>>(h2, glob);
  nbr_kernel  <<<dim3(8, 2),    256, 0, stream>>>(finePts, nbr);
  local_kernel<<<dim3(2048, 2), 256, 0, stream>>>(h2, nbr, locf);

  stage2_kernel<<<dim3(64, 2), 256, 163840, stream>>>(
      h2, locf, glob, w2aF, b2a, w2bF, b2b, w2cF, b2c, wx2F, bx2,
      finePts, secondPts);
}
